// DeepFactor_25683904430333
// MI455X (gfx1250) — matrix operand order repaired and checked by reading, not compiled, not run on hardware
//
#include <hip/hip_runtime.h>
#include <math.h>
#include <stdint.h>

typedef unsigned short u16;
typedef __attribute__((ext_vector_type(16))) __bf16 v16bf;
typedef __attribute__((ext_vector_type(8)))  __bf16 v8bf;
typedef __attribute__((ext_vector_type(8)))  float  v8f;

// ---- CDNA5 async global->LDS path (guarded; falls back to load/ds_store) ----
#if defined(__has_builtin)
#  if __has_builtin(__builtin_amdgcn_global_load_async_to_lds_b128)
#    define USE_ASYNC_LDS 1
#  endif
#  if __has_builtin(__builtin_amdgcn_s_wait_asynccnt)
#    define HAVE_WAIT_ASYNCCNT 1
#  endif
#endif

// Parameter type leaked by the compiler diagnostic: generic pointer to int4-vector.
typedef int v4i_ld __attribute__((vector_size(16)));

__device__ __forceinline__ void wait_async0() {
#ifdef USE_ASYNC_LDS
#  ifdef HAVE_WAIT_ASYNCCNT
    __builtin_amdgcn_s_wait_asynccnt(0);
#  else
    asm volatile("s_wait_asynccnt 0" ::: "memory");
#  endif
#endif
}

// ---------- bf16 helpers (bit-level) ----------
__device__ __forceinline__ u16 f2bf(float f) {
    union { float f; unsigned int u; } v; v.f = f;
    unsigned int r = v.u + 0x7FFFu + ((v.u >> 16) & 1u);
    return (u16)(r >> 16);
}
__device__ __forceinline__ float bf2f(u16 h) {
    union { float f; unsigned int u; } v; v.u = ((unsigned int)h) << 16;
    return v.f;
}

// =====================================================================================
// Generic bf16 WMMA GEMM:  C[m][n] = alpha * sum_k A[m][k]*Bt[n][k]  (+ beta*Dbf[m][n])
// A: (M x K) bf16 row-major.  Bt: (N x K) bf16 row-major (B transposed).
// Double-buffered LDS staging via GLOBAL_LOAD_ASYNC_TO_LDS_B128 (ASYNCcnt) when available.
// M,N multiples of 128, K of 32.
// =====================================================================================
#define GLDS 40  // padded k-stride in LDS (shorts)

__global__ __launch_bounds__(256)
void gemm_bf16_kernel(const u16* __restrict__ A, const u16* __restrict__ Bt,
                      const u16* __restrict__ Dbf, u16* __restrict__ outB,
                      float* __restrict__ outF,
                      int M, int N, int K, float alpha, float beta)
{
    __shared__ __attribute__((aligned(16))) u16 As[2][128 * GLDS];
    __shared__ __attribute__((aligned(16))) u16 Bs[2][128 * GLDS];

    const int tid  = threadIdx.x;
    const int lane = tid & 31;
    const int wave = tid >> 5;
    const int m0   = blockIdx.y * 128;
    const int n0   = blockIdx.x * 128;
    const int wr   = (wave >> 2) * 64;   // wave row offset: 0 / 64
    const int wc   = (wave & 3) * 32;    // wave col offset: 0/32/64/96
    const int lrow = lane & 15;
    const int hi   = lane >> 4;          // half-wave id

    // Stage one 128x32 bf16 tile of A and Bt into LDS buffer `buf`.
    auto stage = [&](int k0, int buf) {
        #pragma unroll
        for (int r = 0; r < 2; ++r) {
            int c   = tid + r * 256;          // 0..511
            int row = c >> 2;
            int ch  = c & 3;
            const u16* ga = &A [(size_t)(m0 + row) * K + k0 + ch * 8];
            const u16* gb = &Bt[(size_t)(n0 + row) * K + k0 + ch * 8];
            u16* la = &As[buf][row * GLDS + ch * 8];
            u16* lb = &Bs[buf][row * GLDS + ch * 8];
#ifdef USE_ASYNC_LDS
            __builtin_amdgcn_global_load_async_to_lds_b128(
                (v4i_ld*)(uintptr_t)ga, (v4i_ld*)(uintptr_t)la, 0, 0);
            __builtin_amdgcn_global_load_async_to_lds_b128(
                (v4i_ld*)(uintptr_t)gb, (v4i_ld*)(uintptr_t)lb, 0, 0);
#else
            *(uint4*)la = *(const uint4*)ga;
            *(uint4*)lb = *(const uint4*)gb;
#endif
        }
    };

    v8f zero = {};
    v8f acc[4][2];
    #pragma unroll
    for (int i = 0; i < 4; ++i)
        #pragma unroll
        for (int j = 0; j < 2; ++j) acc[i][j] = zero;

    const int nk = K >> 5;
    stage(0, 0);
    wait_async0();
    __syncthreads();

    for (int kk = 0; kk < nk; ++kk) {
        const int buf = kk & 1;
        if (kk + 1 < nk) stage((kk + 1) << 5, buf ^ 1);  // overlap fetch with WMMA

        // Fragments. A 16x32 bf16: lanes<16 -> K 0..7 & 16..23; lanes>=16 -> K 8..15 & 24..31
        v16bf af[4], bfr[2];
        #pragma unroll
        for (int ti = 0; ti < 4; ++ti) {
            union { v16bf v; v8bf h[2]; } u;
            const u16* p = &As[buf][(wr + ti * 16 + lrow) * GLDS + hi * 8];
            u.h[0] = *(const v8bf*)(p);
            u.h[1] = *(const v8bf*)(p + 16);
            af[ti] = u.v;
        }
        // B 32x16: lane = column; lanes<16 -> K 0..7 & 16..23; lanes>=16 -> K 8..15 & 24..31
        #pragma unroll
        for (int tj = 0; tj < 2; ++tj) {
            union { v16bf v; v8bf h[2]; } u;
            const u16* p = &Bs[buf][(wc + tj * 16 + lrow) * GLDS + hi * 8];
            u.h[0] = *(const v8bf*)(p);
            u.h[1] = *(const v8bf*)(p + 16);
            bfr[tj] = u.v;
        }
        #pragma unroll
        for (int ti = 0; ti < 4; ++ti)
            #pragma unroll
            for (int tj = 0; tj < 2; ++tj)
                acc[ti][tj] = __builtin_amdgcn_wmma_f32_16x16x32_bf16(
                    false, af[ti], false, bfr[tj], (short)0, acc[ti][tj], false, false);

        if (kk + 1 < nk) wait_async0();  // own async writes done before barrier
        __syncthreads();
    }

    // Epilogue. C layout: VGPR i -> M = i (+8 for upper half-wave), N = lane&15.
    #pragma unroll
    for (int ti = 0; ti < 4; ++ti) {
        #pragma unroll
        for (int tj = 0; tj < 2; ++tj) {
            int mb = m0 + wr + ti * 16 + hi * 8;
            int n  = n0 + wc + tj * 16 + lrow;
            #pragma unroll
            for (int i = 0; i < 8; ++i) {
                size_t off = (size_t)(mb + i) * N + n;
                float  rv  = alpha * acc[ti][tj][i];
                if (Dbf)  rv += beta * bf2f(Dbf[off]);
                if (outF) outF[off] = rv;
                if (outB) outB[off] = f2bf(rv);
            }
        }
    }
}

// =====================================================================================
// Persistent LSTM: 4 workgroups x 16 batch rows, 512 threads (16 waves).
// Per step: z(16x1024) = [x_t | h] (16x320) @ Wt^T via WMMA; gates in VALU; c in regs.
// =====================================================================================
#define XH_S 336   // 320 + pad (shorts)
#define Z_S  1040  // 1024 + pad (shorts)

__global__ __launch_bounds__(512)
void lstm_kernel(const float* __restrict__ x,    // (64,1024,64) fp32
                 const u16*  __restrict__ Wt,    // (1024,320) bf16 = [Wx;Wh]^T
                 const float* __restrict__ bias, // (1024) fp32
                 u16* __restrict__ hOut)         // (64,1024,256) bf16
{
    __shared__ __attribute__((aligned(16))) u16 xh[16 * XH_S];
    __shared__ __attribute__((aligned(16))) u16 z [16 * Z_S];

    const int tid  = threadIdx.x;
    const int lane = tid & 31;
    const int wave = tid >> 5;
    const int bb   = blockIdx.x * 16;
    const int lrow = lane & 15;
    const int hi   = lane >> 4;
    const int n0w  = wave * 64;

    for (int i = tid; i < 16 * XH_S; i += 512) xh[i] = 0;  // h0 = 0
    __syncthreads();
    for (int i = tid; i < 16 * 64; i += 512) {             // load x_0
        int r = i >> 6, f = i & 63;
        xh[r * XH_S + f] = f2bf(x[(size_t)(bb + r) * 65536 + f]);
    }

    float creg[8];
    #pragma unroll
    for (int e = 0; e < 8; ++e) creg[e] = 0.f;

    float bv[4];
    #pragma unroll
    for (int tj = 0; tj < 4; ++tj) bv[tj] = bias[n0w + tj * 16 + lrow];

    v8f zero = {};
    for (int t = 0; t < 1024; ++t) {
        __syncthreads();  // xh (x_t, h_{t-1}) ready
        v8f acc[4];
        #pragma unroll
        for (int tj = 0; tj < 4; ++tj) acc[tj] = zero;

        #pragma unroll
        for (int k0 = 0; k0 < 320; k0 += 32) {
            union { v16bf v; v8bf h[2]; } ua;
            const u16* pa = &xh[lrow * XH_S + k0 + hi * 8];
            ua.h[0] = *(const v8bf*)(pa);
            ua.h[1] = *(const v8bf*)(pa + 16);
            #pragma unroll
            for (int tj = 0; tj < 4; ++tj) {
                union { v16bf v; v8bf h[2]; } ub;
                const u16* pb = &Wt[(size_t)(n0w + tj * 16 + lrow) * 320 + k0 + hi * 8];
                ub.h[0] = *(const v8bf*)(pb);
                ub.h[1] = *(const v8bf*)(pb + 16);
                acc[tj] = __builtin_amdgcn_wmma_f32_16x16x32_bf16(
                    false, ua.v, false, ub.v, (short)0, acc[tj], false, false);
            }
        }
        // z (+bias) -> LDS as bf16
        #pragma unroll
        for (int tj = 0; tj < 4; ++tj) {
            int n = n0w + tj * 16 + lrow;
            #pragma unroll
            for (int i = 0; i < 8; ++i)
                z[(hi * 8 + i) * Z_S + n] = f2bf(acc[tj][i] + bv[tj]);
        }
        __syncthreads();  // z ready; all xh reads done

        // Gates: thread owns 8 fixed (row,unit) pairs; c stays in registers.
        #pragma unroll
        for (int e = 0; e < 8; ++e) {
            int idx = tid * 8 + e;
            int r = idx >> 8, uu = idx & 255;
            float zi = bf2f(z[r * Z_S + uu]);
            float zf = bf2f(z[r * Z_S + 256 + uu]);
            float zg = bf2f(z[r * Z_S + 512 + uu]);
            float zo = bf2f(z[r * Z_S + 768 + uu]);
            float ig = 1.f / (1.f + __expf(-zi));
            float fg = 1.f / (1.f + __expf(-zf));
            float e2g = __expf(2.f * zg);
            float gg = (e2g - 1.f) / (e2g + 1.f);
            float og = 1.f / (1.f + __expf(-zo));
            float c = fg * creg[e] + ig * gg;
            creg[e] = c;
            float e2c = __expf(2.f * c);
            float h  = og * (e2c - 1.f) / (e2c + 1.f);
            u16 hb = f2bf(h);
            xh[r * XH_S + 64 + uu] = hb;
            hOut[((size_t)(bb + r) * 1024 + t) * 256 + uu] = hb;
        }
        if (t + 1 < 1024) {  // stage x_{t+1}
            for (int i = tid; i < 16 * 64; i += 512) {
                int r = i >> 6, f = i & 63;
                xh[r * XH_S + f] = f2bf(x[(size_t)(bb + r) * 65536 + (size_t)(t + 1) * 64 + f]);
            }
        }
    }
}

// =====================================================================================
// Elementwise / layout kernels
// =====================================================================================
__global__ void build_gp_kernel(u16* Kbf, u16* Abf, u16* X0, float c0) {
    size_t i = (size_t)blockIdx.x * 256 + threadIdx.x;  // 1024*1024
    int row = (int)(i >> 10), col = (int)(i & 1023);
    float d = (float)(row - col) * 0.1f;
    float v = __expf(-0.5f * d * d);
    float a = v + (row == col ? 0.1f : 0.0f);
    Kbf[i] = f2bf(v);
    Abf[i] = f2bf(a);
    X0[i]  = f2bf(a * c0);
}

__global__ void prepw_kernel(const float* Wx, const float* Wh, u16* Wt, int total) {
    int i = blockIdx.x * 256 + threadIdx.x;  // 1024*320
    if (i >= total) return;
    int n = i / 320, k = i % 320;
    float v = (k < 64) ? Wx[(size_t)k * 1024 + n] : Wh[(size_t)(k - 64) * 1024 + n];
    Wt[i] = f2bf(v);
}

__global__ void xt_kernel(const float* x, u16* XT) {
    size_t i = (size_t)blockIdx.x * 256 + threadIdx.x;  // 4096*1024
    int n = (int)(i >> 10), t = (int)(i & 1023);
    int b = n >> 6, f = n & 63;
    XT[i] = f2bf(x[(size_t)b * 65536 + (size_t)t * 64 + f]);
}

__global__ void gdense_kernel(const u16* h, const float* Wd, const float* bd, float* g) {
    int i = blockIdx.x * 256 + threadIdx.x;  // 65536 = B*T
    int b = i >> 10, t = i & 1023;
    const u16* hp = &h[((size_t)b * 1024 + t) * 256];
    float s = 0.f;
    #pragma unroll 8
    for (int u = 0; u < 256; ++u) s += bf2f(hp[u]) * Wd[u];
    g[i] = s + bd[0];
}

// u^T[n][t] = g[b,t]/64 + r[t][n]   (attention collapses to g/K exactly)
__global__ void ubuild_kernel(const float* g, const float* C1, u16* uT) {
    size_t i = (size_t)blockIdx.x * 256 + threadIdx.x;  // 4096*1024
    int n = (int)(i >> 10), t = (int)(i & 1023);
    int b = n >> 6;
    uT[i] = f2bf(g[b * 1024 + t] * 0.015625f + C1[(size_t)t * 4096 + n]);
}

__global__ void out_kernel(const float* C2, float* out) {
    size_t i = (size_t)blockIdx.x * 256 + threadIdx.x;  // 64*1024*64
    int b = (int)(i >> 16), t = (int)((i >> 6) & 1023), k = (int)(i & 63);
    out[i] = C2[(size_t)t * 4096 + b * 64 + k];
}

// =====================================================================================
// Host orchestration
// =====================================================================================
static inline void launch_gemm(const void* A, const void* Bt, const void* Dbf,
                               void* outB, void* outF, int M, int N, int K,
                               float alpha, float beta, hipStream_t s) {
    dim3 grid(N / 128, M / 128), blk(256);
    gemm_bf16_kernel<<<grid, blk, 0, s>>>((const u16*)A, (const u16*)Bt, (const u16*)Dbf,
                                          (u16*)outB, (float*)outF, M, N, K, alpha, beta);
}

extern "C" void kernel_launch(void* const* d_in, const int* in_sizes, int n_in,
                              void* d_out, int out_size, void* d_ws, size_t ws_size,
                              hipStream_t stream) {
    const float* x    = (const float*)d_in[0];  // (64,1024,64)
    const float* Wx   = (const float*)d_in[1];  // (64,1024)
    const float* Wh   = (const float*)d_in[2];  // (256,1024)
    const float* bias = (const float*)d_in[3];  // (1024)
    const float* Wd   = (const float*)d_in[4];  // (256)
    const float* bd   = (const float*)d_in[5];  // (1)
    float* out = (float*)d_out;

    char* ws = (char*)d_ws;
    size_t off = 0;
    auto alloc = [&](size_t bytes) { void* p = ws + off; off = (off + bytes + 255) & ~(size_t)255; return p; };
    u16*   Wt  = (u16*)  alloc((size_t)1024 * 320 * 2);   // LSTM weights, transposed bf16
    u16*   Kbf = (u16*)  alloc((size_t)1024 * 1024 * 2);  // RBF kernel matrix
    u16*   Abf = (u16*)  alloc((size_t)1024 * 1024 * 2);  // K + 0.1 I
    u16*   Xa  = (u16*)  alloc((size_t)1024 * 1024 * 2);  // Newton-Schulz ping
    u16*   Xb  = (u16*)  alloc((size_t)1024 * 1024 * 2);  // Newton-Schulz pong
    u16*   T1  = (u16*)  alloc((size_t)1024 * 1024 * 2);  // A @ X
    u16*   Mbf = (u16*)  alloc((size_t)1024 * 1024 * 2);  // K @ A^-1 (symmetric)
    u16*   XT  = (u16*)  alloc((size_t)4096 * 1024 * 2);  // inputs^T bf16 (n=b*64+f, t)
    u16*   hB  = (u16*)  alloc((size_t)64 * 1024 * 256 * 2); // LSTM hidden, bf16
    float* g   = (float*)alloc((size_t)64 * 1024 * 4);    // dense output
    float* C1  = (float*)alloc((size_t)1024 * 4096 * 4);  // r = M @ X
    u16*   uT  = (u16*)  alloc((size_t)4096 * 1024 * 2);  // (f + r)^T bf16
    float* C2  = (float*)alloc((size_t)1024 * 4096 * 4);  // z = M @ u
    (void)ws_size; (void)n_in; (void)in_sizes; (void)out_size;

    // 1) Data-independent GP matrices: K, A = K + 0.1I, X0 = A / ||A||^2-bound
    build_gp_kernel<<<4096, 256, 0, stream>>>(Kbf, Abf, Xa, 1.0f / 680.0f);

    // 2) Newton-Schulz: X' = 2X - X(AX).  All iterates symmetric -> row-major == B^T.
    u16* Xc = Xa; u16* Xn = Xb;
    for (int it = 0; it < 20; ++it) {
        launch_gemm(Abf, Xc, nullptr, T1, nullptr, 1024, 1024, 1024, 1.0f, 0.0f, stream);
        launch_gemm(Xc, T1, Xc, Xn, nullptr, 1024, 1024, 1024, -1.0f, 2.0f, stream);
        u16* tmp = Xc; Xc = Xn; Xn = tmp;
    }
    // 3) M = K @ A^-1  (symmetric)
    launch_gemm(Kbf, Xc, nullptr, Mbf, nullptr, 1024, 1024, 1024, 1.0f, 0.0f, stream);

    // 4) LSTM path
    prepw_kernel<<<1280, 256, 0, stream>>>(Wx, Wh, Wt, 1024 * 320);
    lstm_kernel<<<4, 512, 0, stream>>>(x, Wt, bias, hB);
    gdense_kernel<<<256, 256, 0, stream>>>(hB, Wd, bd, g);

    // 5) r = M @ inputs  (per-series GP smoothing), as C1[t][b*64+f]
    xt_kernel<<<16384, 256, 0, stream>>>(x, XT);
    launch_gemm(Mbf, XT, nullptr, nullptr, C1, 1024, 4096, 1024, 1.0f, 0.0f, stream);

    // 6) u = g/64 + r (transposed, bf16);  z = M @ u
    ubuild_kernel<<<16384, 256, 0, stream>>>(g, C1, uT);
    launch_gemm(Mbf, uT, nullptr, nullptr, C2, 1024, 4096, 1024, 1.0f, 0.0f, stream);

    // 7) (B,T,K) output
    out_kernel<<<16384, 256, 0, stream>>>(C2, out);
}
